// ScaledDotProductAttention_58780922413823
// MI455X (gfx1250) — matrix operand order repaired and checked by reading, not compiled, not run on hardware
//
#include <hip/hip_runtime.h>
#include <stdint.h>

// ---- problem constants (match reference) ----
#define BATCH 32
#define SEQ   2048
#define DIM   64
#define QTILE 64      // q rows per block (16 per wave, 4 waves)
#define KTILE 32      // keys per inner iteration (two 16-key WMMA subtiles)

// LDS row paddings (in halves): every ds_load_b128 16B-aligned, half-wave
// lanes land on disjoint bank groups.
#define KS_PAD 72     // 144 B/row
#define VT_PAD 40     // 80 B/row
#define P_PAD  40     // 80 B/row

typedef _Float16 v16h __attribute__((ext_vector_type(16)));
typedef _Float16 v8h  __attribute__((ext_vector_type(8)));
typedef _Float16 v4h  __attribute__((ext_vector_type(4)));
typedef _Float16 v2h  __attribute__((ext_vector_type(2)));
typedef __fp16   f16x2 __attribute__((ext_vector_type(2)));
typedef float    v8f  __attribute__((ext_vector_type(8)));

#define LOG2E 1.44269504088896340736f

// packed f32x2 -> f16x2 (v_cvt_pk_rtz_f16_f32), bit-cast to _Float16 vector
__device__ __forceinline__ v2h pk_cvt(float a, float b) {
    f16x2 t = __builtin_amdgcn_cvt_pkrtz(a, b);
    return __builtin_bit_cast(v2h, t);
}

// 8 consecutive fp32 -> 8 halves via packed converts (2 global_load_b128)
__device__ __forceinline__ v8h load_cvt8(const float* __restrict__ p) {
    float4 a = ((const float4*)p)[0];
    float4 b = ((const float4*)p)[1];
    v2h p0 = pk_cvt(a.x, a.y);
    v2h p1 = pk_cvt(a.z, a.w);
    v2h p2 = pk_cvt(b.x, b.y);
    v2h p3 = pk_cvt(b.z, b.w);
    v4h q0 = __builtin_shufflevector(p0, p1, 0, 1, 2, 3);
    v4h q1 = __builtin_shufflevector(p2, p3, 0, 1, 2, 3);
    return __builtin_shufflevector(q0, q1, 0, 1, 2, 3, 4, 5, 6, 7);
}

#define CAT16(lo, hi) __builtin_shufflevector((lo), (hi), 0,1,2,3,4,5,6,7,8,9,10,11,12,13,14,15)

// ---- DPP16 16-lane butterflies (replace ds_bpermute shuffles; all four
// patterns permute strictly within a 16-lane row) ----
template <int CTRL>
__device__ __forceinline__ float dpp_movf(float x) {
    return __builtin_bit_cast(float,
        __builtin_amdgcn_update_dpp(0, __builtin_bit_cast(int, x),
                                    CTRL, 0xF, 0xF, true));
}
__device__ __forceinline__ float red16_max(float x) {
    x = fmaxf(x, dpp_movf<0xB1>(x));    // quad_perm(1,0,3,2)  : xor 1
    x = fmaxf(x, dpp_movf<0x4E>(x));    // quad_perm(2,3,0,1)  : xor 2
    x = fmaxf(x, dpp_movf<0x141>(x));   // row_half_mirror     : xor 7
    x = fmaxf(x, dpp_movf<0x140>(x));   // row_mirror          : xor 15
    return x;
}
__device__ __forceinline__ float red16_sum(float x) {
    x += dpp_movf<0xB1>(x);
    x += dpp_movf<0x4E>(x);
    x += dpp_movf<0x141>(x);
    x += dpp_movf<0x140>(x);
    return x;
}

__global__ __launch_bounds__(128)
void fa_fwd_kernel(const float* __restrict__ q, const float* __restrict__ k,
                   const float* __restrict__ v, const uint8_t* __restrict__ mask,
                   float* __restrict__ out)
{
    __shared__ _Float16 Ks[2][KTILE][KS_PAD];   // double-buffered K tile
    __shared__ _Float16 VT[2][DIM][VT_PAD];     // double-buffered V^T tile
    __shared__ _Float16 Pb[4][16][P_PAD];       // per-wave P transpose buffer

    const int tid  = threadIdx.x;
    const int wave = tid >> 5;
    const int lane = tid & 31;
    const int g    = lane >> 4;   // half-wave group
    const int m    = lane & 15;

    const int b     = blockIdx.x >> 5;            // SEQ/QTILE == 32 tiles
    const int qt    = blockIdx.x & 31;
    const int qbase = qt * QTILE + wave * 16;     // first q row of this wave

    // ---- Q fragments (A layout 16x32 f16): lane holds row m,
    // halves 0..7 -> cols 8g..8g+7, halves 8..15 -> cols 16+8g..23+8g ----
    const float* qrow = q + ((size_t)b * SEQ + qbase + m) * DIM;
    v16h Qf[2];
#pragma unroll
    for (int c = 0; c < 2; ++c) {
        v8h lo = load_cvt8(qrow + 32 * c +      8 * g);
        v8h hi = load_cvt8(qrow + 32 * c + 16 + 8 * g);
        Qf[c] = CAT16(lo, hi);
    }

    // ---- accumulators: O = 4 C-frags (rows v+8g, cols 16c+m); m/l per row,
    // softmax state kept in log2 domain ----
    v8f O[4];
#pragma unroll
    for (int c = 0; c < 4; ++c)
#pragma unroll
        for (int i = 0; i < 8; ++i) O[c][i] = 0.0f;

    float mrow[8], lrow[8];
#pragma unroll
    for (int i = 0; i < 8; ++i) { mrow[i] = -__builtin_inff(); lrow[i] = 0.0f; }

    const uint8_t* mrow_base = mask + ((size_t)b * SEQ + qbase) * (size_t)SEQ;
    const float    score_scale = 0.125f * LOG2E;   // 1/sqrt(64) * log2(e)

    int pb = 0;
    for (int jt = 0; jt < SEQ; jt += KTILE, pb ^= 1) {
        // ---- cooperative staging into buffer pb (written at iter i, re-written
        // at iter i+2: the one intervening barrier orders it against iter-i reads) ----
        {
            const int    kk = tid & 31;            // key within tile
            const int    cg = tid >> 5;            // 16-col group of head dim
            const size_t rb = ((size_t)b * SEQ + jt + kk) * DIM + 16 * cg;

            v8h k0 = load_cvt8(k + rb);
            v8h k1 = load_cvt8(k + rb + 8);
            *(v8h*)&Ks[pb][kk][16 * cg]     = k0;
            *(v8h*)&Ks[pb][kk][16 * cg + 8] = k1;

            float4 v0 = ((const float4*)(v + rb))[0];
            float4 v1 = ((const float4*)(v + rb))[1];
            float4 v2 = ((const float4*)(v + rb))[2];
            float4 v3 = ((const float4*)(v + rb))[3];
            float vals[16] = { v0.x, v0.y, v0.z, v0.w, v1.x, v1.y, v1.z, v1.w,
                               v2.x, v2.y, v2.z, v2.w, v3.x, v3.y, v3.z, v3.w };
#pragma unroll
            for (int i = 0; i < 8; ++i) {
                v2h pp = pk_cvt(vals[2 * i], vals[2 * i + 1]);
                VT[pb][16 * cg + 2 * i][kk]     = pp[0];
                VT[pb][16 * cg + 2 * i + 1][kk] = pp[1];
            }
        }
        __syncthreads();

        // ---- S = Q @ K^T for two 16-key subtiles ----
        v8f Sf[2];
#pragma unroll
        for (int sub = 0; sub < 2; ++sub) {
            v8f acc;
#pragma unroll
            for (int i = 0; i < 8; ++i) acc[i] = 0.0f;
#pragma unroll
            for (int c = 0; c < 2; ++c) {
                // B frag of K^T: lane = key column (key 16*sub+m),
                // halves 0..7 -> K-dim rows 8g..8g+7, halves 8..15 -> rows 16+8g..23+8g
                const v8h* bp = (const v8h*)&Ks[pb][16 * sub + m][32 * c + 8 * g];
                const v8h* bq = (const v8h*)&Ks[pb][16 * sub + m][32 * c + 16 + 8 * g];
                v16h Bf = CAT16(bp[0], bq[0]);
                acc = __builtin_amdgcn_wmma_f32_16x16x32_f16(
                          false, Qf[c], false, Bf, (short)0, acc, false, false);
            }
            Sf[sub] = acc;
        }

        // ---- scale (log2 domain) + mask; C layout element =
        // [row vv+8g, key jt+16*sub+m] ----
#pragma unroll
        for (int sub = 0; sub < 2; ++sub)
#pragma unroll
            for (int vv = 0; vv < 8; ++vv) {
                float s = Sf[sub][vv] * score_scale;
                uint8_t mk = mrow_base[(size_t)(vv + 8 * g) * SEQ + jt + 16 * sub + m];
                Sf[sub][vv] = mk ? -__builtin_inff() : s;
            }

        // ---- online softmax (base-2): rows live in 16-lane halves ----
#pragma unroll
        for (int vv = 0; vv < 8; ++vv) {
            float rmax  = red16_max(fmaxf(Sf[0][vv], Sf[1][vv]));
            float mnew  = fmaxf(mrow[vv], rmax);
            float alpha = __builtin_amdgcn_exp2f(mrow[vv] - mnew);
            mrow[vv] = mnew;
            float p0 = __builtin_amdgcn_exp2f(Sf[0][vv] - mnew);
            float p1 = __builtin_amdgcn_exp2f(Sf[1][vv] - mnew);
            Sf[0][vv] = p0; Sf[1][vv] = p1;
            float rs = red16_sum(p0 + p1);
            lrow[vv] = lrow[vv] * alpha + rs;
#pragma unroll
            for (int c = 0; c < 4; ++c) O[c][vv] *= alpha;
        }

        // ---- C-layout -> A-layout transpose of P through per-wave LDS ----
#pragma unroll
        for (int sub = 0; sub < 2; ++sub)
#pragma unroll
            for (int vv = 0; vv < 8; vv += 2) {
                v2h pp = pk_cvt(Sf[sub][vv], Sf[sub][vv + 1]);
                Pb[wave][vv     + 8 * g][16 * sub + m] = pp[0];
                Pb[wave][vv + 1 + 8 * g][16 * sub + m] = pp[1];
            }

        __asm__ volatile("s_wait_dscnt 0" ::: "memory");   // wave-private buffer

        const v8h* pr0 = (const v8h*)&Pb[wave][m][8 * g];
        const v8h* pr1 = (const v8h*)&Pb[wave][m][16 + 8 * g];
        v16h Ap = CAT16(pr0[0], pr1[0]);

        // ---- O += P @ V (4 output column groups) ----
#pragma unroll
        for (int c = 0; c < 4; ++c) {
            const v8h* bv = (const v8h*)&VT[pb][16 * c + m][8 * g];
            const v8h* bw = (const v8h*)&VT[pb][16 * c + m][16 + 8 * g];
            v16h Bv = CAT16(bv[0], bw[0]);
            O[c] = __builtin_amdgcn_wmma_f32_16x16x32_f16(
                       false, Ap, false, Bv, (short)0, O[c], false, false);
        }
    }

    // ---- epilogue: normalize and store ----
    float* orow = out + ((size_t)b * SEQ + qbase) * DIM;
#pragma unroll
    for (int vv = 0; vv < 8; ++vv) {
        float inv = 1.0f / lrow[vv];
#pragma unroll
        for (int c = 0; c < 4; ++c)
            orow[(size_t)(vv + 8 * g) * DIM + 16 * c + m] = O[c][vv] * inv;
    }
}

extern "C" void kernel_launch(void* const* d_in, const int* in_sizes, int n_in,
                              void* d_out, int out_size, void* d_ws, size_t ws_size,
                              hipStream_t stream) {
    const float*   q    = (const float*)d_in[0];
    const float*   k    = (const float*)d_in[1];
    const float*   v    = (const float*)d_in[2];
    const uint8_t* mask = (const uint8_t*)d_in[3];   // numpy/jax bool_ = 1 byte
    float*         out  = (float*)d_out;

    dim3 grid(BATCH * (SEQ / QTILE));   // 32 * 32 = 1024 workgroups
    dim3 block(128);                    // 4 wave32 waves
    fa_fwd_kernel<<<grid, block, 0, stream>>>(q, k, v, mask, out);
    (void)in_sizes; (void)n_in; (void)out_size; (void)d_ws; (void)ws_size;
}
